// SoftQuantizer_84155589198327
// MI455X (gfx1250) — compile-verified, matrix operand order checked by reading, not run on hardware
//
#include <hip/hip_runtime.h>

// ---------------------------------------------------------------------------
// SoftQuantizer for MI455X (gfx1250, wave32, WMMA).
//   w = softmax( exp(-(|z|^2+|e|^2-2 z.e)/gamma) / T , axis=n_e )
// GEMM (32768x1024x256) via v_wmma_f32_16x16x32_bf16. emb is pre-converted to
// bf16 in d_ws (512KB, L2-resident) so B fragments are raw global_load_b128s
// with no conversion VALU; only the single A fragment per k-step is converted
// in-loop via hardware v_cvt (co-executes with the XDL WMMA pipe).
// Softmax fused per 32-row block: 128MB output written once (~7us HBM floor).
// ---------------------------------------------------------------------------

#define E_DIM     256
#define N_E       1024
#define INV_GAMMA 10.0f   // 1/0.1
#define INV_T     10.0f   // 1/0.1
#define TILE_M    32
#define BLOCK_THREADS 512 // 16 waves: 2 row-groups x 8 col-groups

typedef __attribute__((ext_vector_type(8)))  __bf16 v8bf;
typedef __attribute__((ext_vector_type(16))) __bf16 v16bf;
typedef __attribute__((ext_vector_type(8)))  float  v8f;

struct Frag2 { v8bf lo, hi; };

__device__ __forceinline__ v16bf combine(v8bf lo, v8bf hi) {
    Frag2 f{lo, hi};
    return __builtin_bit_cast(v16bf, f);
}

__device__ __forceinline__ v8bf cvt8(float4 a, float4 b) {
    v8bf v;
    v[0] = (__bf16)a.x; v[1] = (__bf16)a.y; v[2] = (__bf16)a.z; v[3] = (__bf16)a.w;
    v[4] = (__bf16)b.x; v[5] = (__bf16)b.y; v[6] = (__bf16)b.z; v[7] = (__bf16)b.w;
    return v;
}

// --- prep: emb f32 -> bf16 (row-major) + |emb_n|^2; one wave32 per row ----
__global__ void emb_prep_kernel(const float* __restrict__ emb,
                                __bf16* __restrict__ ebf,
                                float* __restrict__ en) {
    int row  = (blockIdx.x * blockDim.x + threadIdx.x) >> 5;
    int lane = threadIdx.x & 31;
    if (row >= N_E) return;
    const float4* src = reinterpret_cast<const float4*>(emb + (size_t)row * E_DIM + lane * 8);
    float4 a = src[0], b = src[1];
    *reinterpret_cast<v8bf*>(ebf + (size_t)row * E_DIM + lane * 8) = cvt8(a, b);
    float s = a.x * a.x + a.y * a.y + a.z * a.z + a.w * a.w
            + b.x * b.x + b.y * b.y + b.z * b.z + b.w * b.w;
    #pragma unroll
    for (int off = 16; off > 0; off >>= 1) s += __shfl_xor(s, off, 32);
    if (lane == 0) en[row] = s;
}

// --- fused distance-GEMM + exp + softmax ----------------------------------
__global__ void __launch_bounds__(BLOCK_THREADS)
soft_quant_kernel(const float* __restrict__ z,
                  const __bf16* __restrict__ ebf,
                  const float* __restrict__ embNorm,
                  float* __restrict__ out) {
    __shared__ float sPart[BLOCK_THREADS];
    __shared__ float sZn[TILE_M];
    __shared__ float sRed[TILE_M][8];
    __shared__ float sRowMax[TILE_M];
    __shared__ float sRowSum[TILE_M];

    const int tid   = threadIdx.x;
    const int lane  = tid & 31;
    const int wave  = tid >> 5;
    const int rg    = wave & 1;   // row group: 16 rows each
    const int cg    = wave >> 1;  // col group: 128 cols each
    const int g     = lane >> 4;  // half-wave
    const int l15   = lane & 15;
    const int mBase   = blockIdx.x * TILE_M;
    const int colBase = cg * 128;

    // ---- z row norms for this block's 32 rows ----
    {
        int r = tid >> 4;
        int c = (tid & 15) * 16;
        const float* zr = z + (size_t)(mBase + r) * E_DIM + c;
        float s = 0.f;
        #pragma unroll
        for (int i = 0; i < 16; ++i) { float v = zr[i]; s += v * v; }
        sPart[tid] = s;
    }
    __syncthreads();
    if (tid < TILE_M) {
        float s = 0.f;
        #pragma unroll
        for (int i = 0; i < 16; ++i) s += sPart[tid * 16 + i];
        sZn[tid] = s;
    }
    __syncthreads();

    // ---- GEMM: acc[t] = z_tile(16x256) . emb_cols(256x16), 8 n-tiles/wave ----
    v8f acc[8];
    #pragma unroll
    for (int t = 0; t < 8; ++t) acc[t] = (v8f){0, 0, 0, 0, 0, 0, 0, 0};

    const int    arow = mBase + rg * 16 + l15;
    const float* zrow = z + (size_t)arow * E_DIM;

    #pragma unroll 2
    for (int kb = 0; kb < E_DIM; kb += 32) {
        // A fragment: lane = row M=l15; K chunks g*8+[0..7] and g*8+16+[0..7]
        const float4* ap = reinterpret_cast<const float4*>(zrow + kb + g * 8);
        v16bf afrag = combine(cvt8(ap[0], ap[1]), cvt8(ap[4], ap[5]));
        #pragma unroll
        for (int t = 0; t < 8; ++t) {
            int col = colBase + t * 16 + l15;   // B col n = emb row n
            const v8bf* bp =
                reinterpret_cast<const v8bf*>(ebf + (size_t)col * E_DIM + kb + g * 8);
            v16bf bfrag = combine(bp[0], bp[2]);   // +16 elements = +2 v8bf
            acc[t] = __builtin_amdgcn_wmma_f32_16x16x32_bf16(
                false, afrag, false, bfrag, (short)0, acc[t], false, false);
        }
    }

    // ---- logits l = exp(-(zn+en-2*dot)*INV_GAMMA)*INV_T, kept in acc ----
    #pragma unroll
    for (int t = 0; t < 8; ++t) {
        int   col = colBase + t * 16 + l15;
        float en  = embNorm[col];
        #pragma unroll
        for (int r = 0; r < 8; ++r) {
            int   ml = rg * 16 + r + 8 * g;    // C layout: VGPR r = rows r / r+8
            float dd = sZn[ml] + en - 2.0f * acc[t][r];
            acc[t][r] = __expf(-dd * INV_GAMMA) * INV_T;
        }
    }

    // ---- softmax row max: tiles -> 16-lane butterfly -> 8 waves via LDS ----
    float vmax[8];
    #pragma unroll
    for (int r = 0; r < 8; ++r) {
        float m = acc[0][r];
        #pragma unroll
        for (int t = 1; t < 8; ++t) m = fmaxf(m, acc[t][r]);
        #pragma unroll
        for (int off = 1; off < 16; off <<= 1) m = fmaxf(m, __shfl_xor(m, off, 32));
        vmax[r] = m;
    }
    if (l15 == 0) {
        #pragma unroll
        for (int r = 0; r < 8; ++r) sRed[rg * 16 + r + 8 * g][cg] = vmax[r];
    }
    __syncthreads();
    if (tid < TILE_M) {
        float m = sRed[tid][0];
        #pragma unroll
        for (int j = 1; j < 8; ++j) m = fmaxf(m, sRed[tid][j]);
        sRowMax[tid] = m;
    }
    __syncthreads();

    // ---- softmax row sum of exp(l - max) ----
    #pragma unroll
    for (int r = 0; r < 8; ++r) {
        int   ml = rg * 16 + r + 8 * g;
        float m  = sRowMax[ml];
        float s  = 0.f;
        #pragma unroll
        for (int t = 0; t < 8; ++t) s += __expf(acc[t][r] - m);
        #pragma unroll
        for (int off = 1; off < 16; off <<= 1) s += __shfl_xor(s, off, 32);
        if (l15 == 0) sRed[ml][cg] = s;
    }
    __syncthreads();
    if (tid < TILE_M) {
        float s = 0.f;
        #pragma unroll
        for (int j = 0; j < 8; ++j) s += sRed[tid][j];
        sRowSum[tid] = s;
    }
    __syncthreads();

    // ---- write w = exp(l - max) / sum ----
    #pragma unroll
    for (int r = 0; r < 8; ++r) {
        int    ml     = rg * 16 + r + 8 * g;
        float  m      = sRowMax[ml];
        float  inv    = 1.0f / sRowSum[ml];
        size_t rowOff = (size_t)(mBase + ml) * N_E;
        #pragma unroll
        for (int t = 0; t < 8; ++t) {
            int col = colBase + t * 16 + l15;
            out[rowOff + col] = __expf(acc[t][r] - m) * inv;
        }
    }
}

extern "C" void kernel_launch(void* const* d_in, const int* in_sizes, int n_in,
                              void* d_out, int out_size, void* d_ws, size_t ws_size,
                              hipStream_t stream) {
    const float* z   = (const float*)d_in[0];   // [32768, 256] f32
    const float* emb = (const float*)d_in[1];   // [1024, 256]  f32
    float*       out = (float*)d_out;           // [32768, 1024] f32

    // workspace layout: [0, 512KB) bf16 emb; then 4KB of |e|^2
    __bf16* ebf = (__bf16*)d_ws;
    float*  en  = (float*)((char*)d_ws + (size_t)N_E * E_DIM * sizeof(unsigned short));

    const int rows = in_sizes[0] / E_DIM;       // 32768

    emb_prep_kernel<<<(N_E * 32 + 255) / 256, 256, 0, stream>>>(emb, ebf, en);
    soft_quant_kernel<<<rows / TILE_M, BLOCK_THREADS, 0, stream>>>(z, ebf, en, out);
}
